// SpatialGatedBlock_670014898399
// MI455X (gfx1250) — compile-verified, matrix operand order checked by reading, not run on hardware
//
#include <hip/hip_runtime.h>
#include <hip/hip_bf16.h>

typedef _Float16 v8h  __attribute__((ext_vector_type(8)));
typedef _Float16 v16h __attribute__((ext_vector_type(16)));
typedef float    v8f  __attribute__((ext_vector_type(8)));

#define D 128

// ---------------------------------------------------------------------------
// Pack a 128x128 row-major f32 W into the WMMA 16-bit B-fragment layout.
// For (kstep kt in 0..3, ntile nt in 0..7): a 32-lane x 16-f16 block, lane-major
// (each lane's 16 f16 = 32 contiguous bytes). Lane L covers col N = 16*nt+(L&15),
// K values {kk..kk+7, kk+16..kk+23} with kk = 32*kt + 8*(L>>4).
// ---------------------------------------------------------------------------
__global__ void pack_w_kernel(const float* __restrict__ W, _Float16* __restrict__ P) {
    int tid  = threadIdx.x;              // 0..1023
    int lane = tid & 31;
    int tile = tid >> 5;                 // kt*8 + nt
    int kt   = tile >> 3, nt = tile & 7;
    int kk   = 32 * kt + 8 * (lane >> 4);
    int col  = 16 * nt + (lane & 15);
    _Float16* dst = P + (size_t)tile * 512 + (size_t)lane * 16;
#pragma unroll
    for (int j = 0; j < 8; ++j) {
        int kb = kk + ((j & 4) ? 16 : 0) + 2 * (j & 3);
        dst[2 * j]     = (_Float16)W[(size_t)kb * D + col];
        dst[2 * j + 1] = (_Float16)W[(size_t)(kb + 1) * D + col];
    }
}

// ---- helpers --------------------------------------------------------------
__device__ __forceinline__ void load_bfrag(v16h b[4], const _Float16* __restrict__ HN,
                                           int row, int half) {
    const _Float16* brow = HN + (size_t)row * D;
#pragma unroll
    for (int kt = 0; kt < 4; ++kt) {
        const int kk = kt * 32 + half * 8;
        v8h lo = *(const v8h*)(brow + kk);
        v8h hi = *(const v8h*)(brow + kk + 16);
#pragma unroll
        for (int i = 0; i < 8; ++i) { b[kt][i] = lo[i]; b[kt][8 + i] = hi[i]; }
    }
}

// 4 WMMAs (K=128) on one 16x16 score tile, then sigmoid + accumulate.
__device__ __forceinline__ void tile_acc(const v16h a[4], const v16h b[4], float acc[8]) {
    v8f c = {};
#pragma unroll
    for (int kt = 0; kt < 4; ++kt)
        c = __builtin_amdgcn_wmma_f32_16x16x32_f16(false, a[kt], false, b[kt], (short)0, c,
                                                   false, false);
#pragma unroll
    for (int r = 0; r < 8; ++r)
        acc[r] += __builtin_amdgcn_rcpf(1.0f + __expf(-c[r]));   // sigmoid
}

// ---------------------------------------------------------------------------
// C[16x128 tile] = X[16x128 rows] @ W (via packed B frags), WMMA f16->f32.
// Block = 256 threads = 8 waves; wave w owns column tile n0 = 16*w.
// NORM=true: row-normalize (cosine prep) and store f16; else store f32.
// ---------------------------------------------------------------------------
template <bool NORM>
__global__ void gemm128_kernel(const float* __restrict__ X,
                               const _Float16* __restrict__ PW,
                               _Float16* __restrict__ outH,
                               float* __restrict__ outF) {
    __shared__ float sumsq[16];
    const int tid  = threadIdx.x;
    const int lane = tid & 31;
    const int w    = __builtin_amdgcn_readfirstlane(tid >> 5);   // wave-uniform ntile
    const int half = lane >> 4;
    const int idx  = lane & 15;
    const int r0   = blockIdx.x * 16;
    const int n0   = w * 16;

    v8f c = {};
#pragma unroll
    for (int kt = 0; kt < 4; ++kt) {
        const int kk = kt * 32 + half * 8;
        const float* ap = X + (size_t)(r0 + idx) * D + kk;
        v16h a;
#pragma unroll
        for (int i = 0; i < 8; ++i) {
            a[i]     = (_Float16)ap[i];
            a[8 + i] = (_Float16)ap[16 + i];
        }
        const v16h b = *(const v16h*)(PW + (size_t)(kt * 8 + w) * 512 + (size_t)lane * 16);
        c = __builtin_amdgcn_wmma_f32_16x16x32_f16(false, a, false, b, (short)0, c, false, false);
    }

    if (NORM) {
        if (tid < 16) sumsq[tid] = 0.0f;
        __syncthreads();
#pragma unroll
        for (int r = 0; r < 8; ++r) atomicAdd(&sumsq[r + half * 8], c[r] * c[r]);
        __syncthreads();
#pragma unroll
        for (int r = 0; r < 8; ++r) {
            float nrm = __builtin_sqrtf(sumsq[r + half * 8]);
            float s   = __builtin_amdgcn_rcpf(fmaxf(nrm, 1e-12f));
            outH[(size_t)(r0 + r + half * 8) * D + n0 + idx] = (_Float16)(c[r] * s);
        }
    } else {
#pragma unroll
        for (int r = 0; r < 8; ++r)
            outF[(size_t)(r0 + r + half * 8) * D + n0 + idx] = c[r];
    }
}

// ---------------------------------------------------------------------------
// att[m] = mean_n sigmoid(hn[m] . hn[n]).  Block = 16 rows (m0), 8 waves
// scalar-strided over the column tiles. B fragments double-buffered so the
// next tile's 8 global_load_b128 overlap the current tile's WMMA+sigmoid.
// ---------------------------------------------------------------------------
__global__ void att_kernel(const _Float16* __restrict__ HN, float* __restrict__ att,
                           int nTiles, float invN) {
    __shared__ float ssum[16];
    const int tid  = threadIdx.x;
    const int lane = tid & 31;
    const int w    = __builtin_amdgcn_readfirstlane(tid >> 5);   // SGPR: scalar loop ctrl
    const int half = lane >> 4;
    const int idx  = lane & 15;
    const int m0   = blockIdx.x * 16;

    // A fragments for all 4 k-steps, held in registers (32 VGPRs).
    v16h a[4];
    load_bfrag(a, HN, m0 + idx, half);

    float acc[8] = {0.f, 0.f, 0.f, 0.f, 0.f, 0.f, 0.f, 0.f};

    v16h b0[4], b1[4];
    int nt = w;
    if (nt < nTiles) load_bfrag(b0, HN, nt * 16 + idx, half);
    for (; nt + 8 < nTiles; nt += 16) {
        load_bfrag(b1, HN, (nt + 8) * 16 + idx, half);           // overlap with b0 compute
        if (nt + 24 < nTiles)                                    // scalar-guarded prefetch
            __builtin_prefetch(HN + (size_t)((nt + 24) * 16 + idx) * D, 0, 1);
        tile_acc(a, b0, acc);
        if (nt + 16 < nTiles)
            load_bfrag(b0, HN, (nt + 16) * 16 + idx, half);      // overlap with b1 compute
        tile_acc(a, b1, acc);
    }
    if (nt < nTiles) tile_acc(a, b0, acc);                       // odd tail

    if (tid < 16) ssum[tid] = 0.0f;
    __syncthreads();
#pragma unroll
    for (int r = 0; r < 8; ++r) atomicAdd(&ssum[r + half * 8], acc[r]);
    __syncthreads();
    if (tid < 16) att[m0 + tid] = ssum[tid] * invN;
}

// --------------------------- GCN scatter path ------------------------------
__global__ void deg_init_kernel(float* __restrict__ deg, int n) {
    int i = blockIdx.x * blockDim.x + threadIdx.x;
    if (i < n) deg[i] = 1.0f;            // self-loop
}

__global__ void deg_count_kernel(const int* __restrict__ dst, float* __restrict__ deg, int e) {
    int i = blockIdx.x * blockDim.x + threadIdx.x;
    if (i < e) atomicAdd(&deg[dst[i]], 1.0f);
}

__global__ void dinv_kernel(float* __restrict__ deg, int n) {
    int i = blockIdx.x * blockDim.x + threadIdx.x;
    if (i < n) {
        float d = deg[i];
        deg[i] = (d > 0.0f) ? __builtin_amdgcn_rsqf(d) : 0.0f;   // in-place -> dinv
    }
}

// g[i][:] = xw[i][:] * dinv[i]^2   (self-loop contribution), into d_out
__global__ void self_loop_kernel(const float* __restrict__ xw, const float* __restrict__ dinv,
                                 float* __restrict__ g, int n) {
    int t = blockIdx.x * blockDim.x + threadIdx.x;
    if (t < n * D) {
        int m = t >> 7;
        float s = dinv[m];
        g[t] = xw[t] * s * s;
    }
}

// one thread per (edge, 4-col group): 4 atomic f32 adds
__global__ void scatter_kernel(const int* __restrict__ src, const int* __restrict__ dst,
                               const float* __restrict__ xw, const float* __restrict__ dinv,
                               float* __restrict__ g, int e) {
    int t = blockIdx.x * blockDim.x + threadIdx.x;
    if (t >= e * (D / 4)) return;
    int ed = t >> 5;                     // D/4 = 32 groups per edge
    int c4 = (t & 31) * 4;
    int s  = src[ed], d2 = dst[ed];
    float wgt = dinv[s] * dinv[d2];
    const float4 v = *(const float4*)(xw + (size_t)s * D + c4);
    float* gp = g + (size_t)d2 * D + c4;
    atomicAdd(gp + 0, v.x * wgt);
    atomicAdd(gp + 1, v.y * wgt);
    atomicAdd(gp + 2, v.z * wgt);
    atomicAdd(gp + 3, v.w * wgt);
}

__global__ void finalize_kernel(float* __restrict__ g, const float* __restrict__ b,
                                const float* __restrict__ att, int n) {
    int t = blockIdx.x * blockDim.x + threadIdx.x;
    if (t < n * D) {
        int m = t >> 7, dcol = t & (D - 1);
        g[t] = (g[t] + b[dcol]) * att[m];
    }
}

// ---------------------------------------------------------------------------
extern "C" void kernel_launch(void* const* d_in, const int* in_sizes, int n_in,
                              void* d_out, int out_size, void* d_ws, size_t ws_size,
                              hipStream_t stream) {
    const float* x     = (const float*)d_in[0];
    const int*   ei    = (const int*)d_in[1];
    const float* W_att = (const float*)d_in[2];
    const float* W_gcn = (const float*)d_in[3];
    const float* b_gcn = (const float*)d_in[4];
    float* out = (float*)d_out;

    const int N = in_sizes[0] / D;       // 12288
    const int E = in_sizes[1] / 2;       // 393216
    const int nTiles = N / 16;           // 768
    const int* src = ei;
    const int* dst = ei + E;

    // workspace layout (256B aligned)
    char* ws = (char*)d_ws;
    _Float16* pWatt = (_Float16*)(ws);                                   //  32 KB
    _Float16* pWgcn = (_Float16*)(ws + 32768);                           //  32 KB
    _Float16* hn    = (_Float16*)(ws + 65536);                           // N*D*2
    float*    xw    = (float*)(ws + 65536 + (size_t)N * D * 2);          // N*D*4
    float*    att   = (float*)((char*)xw + (size_t)N * D * 4);           // N*4
    float*    dinv  = (float*)((char*)att + (size_t)N * 4);              // N*4

    // 1) pack both weight matrices into WMMA-B fragment layout
    pack_w_kernel<<<1, 1024, 0, stream>>>(W_att, pWatt);
    pack_w_kernel<<<1, 1024, 0, stream>>>(W_gcn, pWgcn);

    // 2) hn = rownorm(x @ W_att) as f16
    gemm128_kernel<true><<<nTiles, 256, 0, stream>>>(x, pWatt, hn, nullptr);

    // 3) att[m] = mean_n sigmoid(hn @ hn^T)
    att_kernel<<<nTiles, 256, 0, stream>>>(hn, att, nTiles, 1.0f / (float)N);

    // 4) xw = x @ W_gcn
    gemm128_kernel<false><<<nTiles, 256, 0, stream>>>(x, pWgcn, nullptr, xw);

    // 5) symmetric-normalized degree
    deg_init_kernel<<<(N + 255) / 256, 256, 0, stream>>>(dinv, N);
    deg_count_kernel<<<(E + 255) / 256, 256, 0, stream>>>(dst, dinv, E);
    dinv_kernel<<<(N + 255) / 256, 256, 0, stream>>>(dinv, N);

    // 6) g = self-loop term, then edge scatter, then (g + b) * att
    self_loop_kernel<<<(N * D + 255) / 256, 256, 0, stream>>>(xw, dinv, out, N);
    scatter_kernel<<<(E * (D / 4) + 255) / 256, 256, 0, stream>>>(src, dst, xw, dinv, out, E);
    finalize_kernel<<<(N * D + 255) / 256, 256, 0, stream>>>(out, b_gcn, att, N);
}
